// GraphSAGE_13245679141137
// MI455X (gfx1250) — matrix operand order repaired and checked by reading, not compiled, not run on hardware
//
#include <hip/hip_runtime.h>
#include <hip/hip_bf16.h>

// ---------------------------------------------------------------------------
// Problem constants
// ---------------------------------------------------------------------------
#define N0_ 200000
#define N1_ 100000
#define N2_ 50000
#define E0_ 1600000
#define E1_ 800000
#define DIN_ 256
#define H_ 256
#define C_ 7
#define K_ 40
#define KTOT_ 512          // concat K for both conv GEMMs and the head GEMM
#define NFIN_ 320          // 7 + 280 = 287 head outputs, padded to 320

// ---- workspace layout: fp32 region, then ushort (bf16) region --------------
#define OFF_SUM0   ((size_t)0)
#define OFF_DEG0   (OFF_SUM0 + (size_t)N1_ * H_)
#define OFF_SUM1   (OFF_DEG0 + (size_t)N1_)
#define OFF_DEG1   (OFF_SUM1 + (size_t)N2_ * H_)
#define OFF_H1     (OFF_DEG1 + (size_t)N2_)
#define OFF_ALLOUT (OFF_H1   + (size_t)N1_ * H_)
#define OFF_BF16   (OFF_ALLOUT + (size_t)N2_ * NFIN_)     // ushort area starts here
// ushort offsets inside bf16 area
#define UO_A0      ((size_t)0)                            // [N1][512]  layer0 A
#define UO_A1      (UO_A0 + (size_t)N1_ * KTOT_)          // [N2][512]  layer1 A
#define UO_HC      (UO_A1 + (size_t)N2_ * KTOT_)          // [N2][512]  head A (hcat)
#define UO_WC0     (UO_HC + (size_t)N2_ * KTOT_)          // [256][512]
#define UO_WC1     (UO_WC0 + (size_t)H_ * KTOT_)          // [256][512]
#define UO_WF      (UO_WC1 + (size_t)H_ * KTOT_)          // [320][512]

typedef __attribute__((ext_vector_type(16))) __bf16 v16bf;
typedef __attribute__((ext_vector_type(8)))  float  v8f;

union Frag16 {
  unsigned u[8];
  v16bf    v;
};

__device__ inline unsigned bf16_rne(float f) {
  unsigned u = __float_as_uint(f);
  return (u + 0x7FFFu + ((u >> 16) & 1u)) >> 16;
}

// ---------------------------------------------------------------------------
// Weight conversion: fp32 -> bf16, [N][K] row-major (K contiguous) B matrices
// ---------------------------------------------------------------------------
__global__ __launch_bounds__(256) void convert_weights_kernel(
    const float* __restrict__ Ws0, const float* __restrict__ Wn0,
    const float* __restrict__ Ws1, const float* __restrict__ Wn1,
    const float* __restrict__ Wpr, const float* __restrict__ Wpp,
    unsigned short* __restrict__ Wc0, unsigned short* __restrict__ Wc1,
    unsigned short* __restrict__ Wf) {
  int tid = blockIdx.x * 256 + threadIdx.x;       // 0 .. 320*512-1
  int k = tid & (KTOT_ - 1);
  int r = tid >> 9;
  if (tid < H_ * KTOT_) {
    float v0 = (k < H_) ? Ws0[r * H_ + k] : Wn0[r * H_ + (k - H_)];
    float v1 = (k < H_) ? Ws1[r * H_ + k] : Wn1[r * H_ + (k - H_)];
    Wc0[tid] = (unsigned short)bf16_rne(v0);
    Wc1[tid] = (unsigned short)bf16_rne(v1);
  }
  float vf = 0.0f;
  if (r < C_)                vf = Wpr[r * KTOT_ + k];
  else if (r < C_ + C_ * K_) vf = Wpp[(size_t)(r - C_) * KTOT_ + k];
  Wf[tid] = (unsigned short)bf16_rne(vf);
}

// inputs[:N1] fp32 -> bf16 into A0 cols [0,256)
__global__ __launch_bounds__(256) void convert_in_kernel(
    const float* __restrict__ x, unsigned short* __restrict__ Ab) {
  size_t i = (size_t)blockIdx.x * 256 + threadIdx.x;   // < N1*256
  size_t n = i >> 8;
  size_t d = i & 255;
  Ab[n * KTOT_ + d] = (unsigned short)bf16_rne(x[i]);
}

// ---------------------------------------------------------------------------
// Edge scatter: one wave per edge, 8 fp32 atomics/lane + degree count
// ---------------------------------------------------------------------------
__global__ __launch_bounds__(256) void scatter_kernel(
    const float* __restrict__ Hsrc, const int* __restrict__ esrc,
    const int* __restrict__ edst, float* __restrict__ sum,
    float* __restrict__ deg, int E) {
  int lane = threadIdx.x & 31;
  int e = blockIdx.x * 8 + (threadIdx.x >> 5);
  if (e >= E) return;
  int s = esrc[e];
  int d = edst[e];
  const float4* src = (const float4*)(Hsrc + (size_t)s * H_) + lane * 2;
  float4 x0 = src[0];
  float4 x1 = src[1];
  float* dst = sum + (size_t)d * H_ + lane * 8;
  atomicAdd(dst + 0, x0.x); atomicAdd(dst + 1, x0.y);
  atomicAdd(dst + 2, x0.z); atomicAdd(dst + 3, x0.w);
  atomicAdd(dst + 4, x1.x); atomicAdd(dst + 5, x1.y);
  atomicAdd(dst + 6, x1.z); atomicAdd(dst + 7, x1.w);
  if (lane == 0) atomicAdd(deg + d, 1.0f);
}

// mean = sum / max(deg,1), written as bf16 into A-buffer cols [256,512)
__global__ __launch_bounds__(256) void mean_bf16_kernel(
    const float* __restrict__ sum, const float* __restrict__ deg,
    unsigned short* __restrict__ Ab) {
  size_t i = (size_t)blockIdx.x * 256 + threadIdx.x;   // < rows*256
  size_t row = i >> 8;
  size_t d = i & 255;
  float m = sum[i] / fmaxf(deg[row], 1.0f);
  Ab[row * KTOT_ + H_ + d] = (unsigned short)bf16_rne(m);
}

// hcat cols [256,512) = bf16(relu(h1[:N2])) == bf16(h1[:N2])
__global__ __launch_bounds__(256) void copy_neighbor_kernel(
    const float* __restrict__ h1, unsigned short* __restrict__ hcatb) {
  size_t i = (size_t)blockIdx.x * 256 + threadIdx.x;   // < N2*256
  size_t n = i >> 8;
  size_t d = i & 255;
  hcatb[n * KTOT_ + H_ + d] = (unsigned short)bf16_rne(h1[i]);
}

// ---------------------------------------------------------------------------
// WMMA bf16 GEMM: C[M x N] = act( A_bf16[M][Ktot] @ W_bf16[N][Ktot]^T + bias )
//   One wave -> 16x64 tile (4 accumulators reusing one A fragment).
//   A fragment = two global_load_b128 per lane (K-pairs {0,2,4,6}/{16..22}
//   + halfsel*8 are contiguous bf16 in memory). B fragment likewise.
//   Epilogue writes optional fp32 C and/or bf16 C (fused downstream convert).
// ---------------------------------------------------------------------------
__global__ __launch_bounds__(256) void gemm_bf16_kernel(
    const unsigned short* __restrict__ Ab, int Ktot,
    const unsigned short* __restrict__ W,
    const float* __restrict__ bias,
    float* __restrict__ Cf, int ldcf,
    unsigned short* __restrict__ Cb, int ldcb, int cbRows,
    int Mtiles, int relu) {
  int lane = threadIdx.x & 31;
  int wave = threadIdx.x >> 5;
  int mtile = blockIdx.y * 8 + wave;
  if (mtile >= Mtiles) return;                 // wave-uniform: EXEC stays all-1s
  int mrow = mtile * 16;
  int nbase = blockIdx.x * 64;
  int halfsel = (lane >> 4) & 1;
  int mlocal = lane & 15;
  int m = mrow + mlocal;

  const unsigned short* Abase = Ab + (size_t)m * Ktot + halfsel * 8;

  v8f acc0 = {}, acc1 = {}, acc2 = {}, acc3 = {};

  for (int kt = 0; kt < Ktot; kt += 32) {
    __builtin_prefetch(Abase + kt + 32, 0, 0);   // global_prefetch_b8, next K step
    Frag16 a;
    uint4 q0 = *(const uint4*)(Abase + kt);        // K pairs 0,2,4,6 (+halfsel*8)
    uint4 q1 = *(const uint4*)(Abase + kt + 16);   // K pairs 16,18,20,22
    a.u[0] = q0.x; a.u[1] = q0.y; a.u[2] = q0.z; a.u[3] = q0.w;
    a.u[4] = q1.x; a.u[5] = q1.y; a.u[6] = q1.z; a.u[7] = q1.w;
#pragma unroll
    for (int j = 0; j < 4; ++j) {
      int ncol = nbase + j * 16 + mlocal;
      const unsigned short* Wp = W + (size_t)ncol * Ktot + kt + halfsel * 8;
      Frag16 b;
      uint4 w0 = *(const uint4*)(Wp);
      uint4 w1 = *(const uint4*)(Wp + 16);
      b.u[0] = w0.x; b.u[1] = w0.y; b.u[2] = w0.z; b.u[3] = w0.w;
      b.u[4] = w1.x; b.u[5] = w1.y; b.u[6] = w1.z; b.u[7] = w1.w;
      if (j == 0)
        acc0 = __builtin_amdgcn_wmma_f32_16x16x32_bf16(false, a.v, false, b.v,
                                                       (short)0, acc0, false, false);
      else if (j == 1)
        acc1 = __builtin_amdgcn_wmma_f32_16x16x32_bf16(false, a.v, false, b.v,
                                                       (short)0, acc1, false, false);
      else if (j == 2)
        acc2 = __builtin_amdgcn_wmma_f32_16x16x32_bf16(false, a.v, false, b.v,
                                                       (short)0, acc2, false, false);
      else
        acc3 = __builtin_amdgcn_wmma_f32_16x16x32_bf16(false, a.v, false, b.v,
                                                       (short)0, acc3, false, false);
    }
  }

  // Epilogue: C layout -> lane&15 = N index, VGPR v = M row (+8 for hi half)
#pragma unroll
  for (int j = 0; j < 4; ++j) {
    v8f acc = (j == 0) ? acc0 : (j == 1) ? acc1 : (j == 2) ? acc2 : acc3;
    int ncol = nbase + j * 16 + mlocal;
    float bv = bias ? bias[ncol] : 0.0f;
#pragma unroll
    for (int v = 0; v < 8; ++v) {
      int row = mrow + v + halfsel * 8;
      float val = acc[v] + bv;
      if (relu) val = fmaxf(val, 0.0f);
      if (Cf) Cf[(size_t)row * ldcf + ncol] = val;
      if (Cb && row < cbRows)
        Cb[(size_t)row * ldcb + ncol] = (unsigned short)bf16_rne(val);
    }
  }
}

// argmax over 7 scores, gather the selected class's 40-dim projection
__global__ __launch_bounds__(256) void select_kernel(
    const float* __restrict__ allout, float* __restrict__ out) {
  int i = blockIdx.x * 256 + threadIdx.x;
  if (i >= N2_ * K_) return;
  int n = i / K_;
  int k = i - n * K_;
  const float* s = allout + (size_t)n * NFIN_;
  int best = 0;
  float bv = s[0];
#pragma unroll
  for (int c = 1; c < C_; ++c) {
    float v = s[c];
    if (v > bv) { bv = v; best = c; }       // strict > keeps first max (jnp.argmax)
  }
  out[i] = s[C_ + best * K_ + k];
}

// ---------------------------------------------------------------------------
extern "C" void kernel_launch(void* const* d_in, const int* in_sizes, int n_in,
                              void* d_out, int out_size, void* d_ws, size_t ws_size,
                              hipStream_t stream) {
  const float* inputs = (const float*)d_in[0];
  const float* Ws0    = (const float*)d_in[1];
  const float* Wn0    = (const float*)d_in[2];
  const float* b0     = (const float*)d_in[3];
  const float* Ws1    = (const float*)d_in[4];
  const float* Wn1    = (const float*)d_in[5];
  const float* b1     = (const float*)d_in[6];
  const float* Wpr    = (const float*)d_in[7];
  const float* Wpp    = (const float*)d_in[8];
  const int*   es0    = (const int*)d_in[9];
  const int*   ed0    = (const int*)d_in[10];
  const int*   es1    = (const int*)d_in[11];
  const int*   ed1    = (const int*)d_in[12];

  float* ws = (float*)d_ws;
  float* sum0   = ws + OFF_SUM0;
  float* deg0   = ws + OFF_DEG0;
  float* sum1   = ws + OFF_SUM1;
  float* deg1   = ws + OFF_DEG1;
  float* h1     = ws + OFF_H1;
  float* allout = ws + OFF_ALLOUT;
  unsigned short* ub   = (unsigned short*)(ws + OFF_BF16);
  unsigned short* A0b  = ub + UO_A0;    // layer0 GEMM A: [inputs | mean0]
  unsigned short* A1b  = ub + UO_A1;    // layer1 GEMM A: [h1[:N2] | mean1]
  unsigned short* HCb  = ub + UO_HC;    // head  GEMM A: [h2 | relu(h1[:N2])]
  unsigned short* Wc0  = ub + UO_WC0;
  unsigned short* Wc1  = ub + UO_WC1;
  unsigned short* Wf   = ub + UO_WF;
  float* out = (float*)d_out;

  // 1) zero the atomic accumulators (sum0|deg0|sum1|deg1 are contiguous)
  size_t zero_floats = (size_t)N1_ * H_ + N1_ + (size_t)N2_ * H_ + N2_;
  hipMemsetAsync(sum0, 0, zero_floats * sizeof(float), stream);

  // 2) weights + inputs -> bf16
  convert_weights_kernel<<<(NFIN_ * KTOT_) / 256, 256, 0, stream>>>(
      Ws0, Wn0, Ws1, Wn1, Wpr, Wpp, Wc0, Wc1, Wf);
  convert_in_kernel<<<(N1_ * H_) / 256, 256, 0, stream>>>(inputs, A0b);

  // 3) layer-0 segment sum + degree -> bf16 mean into A0b cols [256,512)
  scatter_kernel<<<E0_ / 8, 256, 0, stream>>>(inputs, es0, ed0, sum0, deg0, E0_);
  mean_bf16_kernel<<<(N1_ * H_) / 256, 256, 0, stream>>>(sum0, deg0, A0b);

  // 4) layer-0 GEMM: h1 = relu(A0b @ Wc0^T + b0); rows<N2 also -> bf16 A1b[:, :256]
  {
    dim3 grid(H_ / 64, (N1_ / 16 + 7) / 8);
    gemm_bf16_kernel<<<grid, 256, 0, stream>>>(
        A0b, KTOT_, Wc0, b0, h1, H_, A1b, KTOT_, N2_, N1_ / 16, 1);
  }

  // 5) layer-1 segment sum + degree -> bf16 mean into A1b cols [256,512)
  scatter_kernel<<<E1_ / 8, 256, 0, stream>>>(h1, es1, ed1, sum1, deg1, E1_);
  mean_bf16_kernel<<<(N2_ * H_) / 256, 256, 0, stream>>>(sum1, deg1, A1b);

  // 6) layer-1 GEMM: HCb[:, :256] = bf16(relu(A1b @ Wc1^T + b1))
  {
    dim3 grid(H_ / 64, (N2_ / 16 + 7) / 8);
    gemm_bf16_kernel<<<grid, 256, 0, stream>>>(
        A1b, KTOT_, Wc1, b1, nullptr, 0, HCb, KTOT_, N2_, N2_ / 16, 1);
  }

  // 7) HCb[:, 256:) = bf16(relu(h1[:N2])) (== bf16(h1[:N2]))
  copy_neighbor_kernel<<<(N2_ * H_) / 256, 256, 0, stream>>>(h1, HCb);

  // 8) head GEMM: allout = HCb @ [W_prompt ; W_pp]_bf16^T   (N padded to 320)
  {
    dim3 grid(NFIN_ / 64, (N2_ / 16 + 7) / 8);
    gemm_bf16_kernel<<<grid, 256, 0, stream>>>(
        HCb, KTOT_, Wf, nullptr, allout, NFIN_, nullptr, 0, 0, N2_ / 16, 0);
  }

  // 9) argmax + gather selected projection -> d_out (N2 x 40)
  select_kernel<<<(N2_ * K_ + 255) / 256, 256, 0, stream>>>(allout, out);
}
